// Qwen3_5MoeGatedDeltaNet_33371895890023
// MI455X (gfx1250) — compile-verified, matrix operand order checked by reading, not run on hardware
//
#include <hip/hip_runtime.h>
#include <cmath>

// ---------------------------------------------------------------------------
// Types for CDNA5 WMMA (gfx1250, wave32)
// ---------------------------------------------------------------------------
typedef __bf16 bf16_t;
typedef bf16_t v16bf __attribute__((ext_vector_type(16)));
typedef float  v8f   __attribute__((ext_vector_type(8)));

#define DN_B      2
#define DN_S      4096
#define DN_HID    2048
#define DN_HV     32
#define DN_HK     16
#define DN_DK     128
#define DN_DV     128
#define DN_CONV   8192
#define DN_KEYDIM 2048
#define DN_VALDIM 4096
#define DN_CHUNK  64
#define DN_NC     64

// ---------------------------------------------------------------------------
// CDNA5 async global->LDS copy (ASYNCcnt path). The builtin's signature (from
// the compiler diagnostic) is:
//   void __builtin_amdgcn_global_load_async_to_lds_b128(
//       v4i addrspace(1)* src, v4i addrspace(3)* dst, imm int off, imm int cpol)
// ---------------------------------------------------------------------------
#if defined(__has_builtin)
#if __has_builtin(__builtin_amdgcn_global_load_async_to_lds_b128)
#define DN_HAVE_ASYNC 1
#endif
#endif

#ifdef DN_HAVE_ASYNC
typedef int dn_v4i __attribute__((vector_size(16)));
typedef __attribute__((address_space(1))) dn_v4i* dn_gptr_t;
typedef __attribute__((address_space(3))) dn_v4i* dn_lptr_t;
static __device__ inline void async_b128(void* lds_dst, const void* gsrc) {
  __builtin_amdgcn_global_load_async_to_lds_b128((dn_gptr_t)gsrc,
                                                 (dn_lptr_t)lds_dst, 0, 0);
}
static __device__ inline void async_wait() {
  asm volatile("s_wait_asynccnt 0x0" ::: "memory");
}
#else
static __device__ inline void async_b128(void* lds_dst, const void* gsrc) {
  *(float4*)lds_dst = *(const float4*)gsrc;
}
static __device__ inline void async_wait() {}
#endif

// ---------------------------------------------------------------------------
// Fragment helpers (per ISA 7.12.2 layouts)
// A-frag 16x32 bf16: lanes 0-15 -> rows, K {kh..kh+7, kh+16..kh+23}, kh = 8*(lane>=16)
// C/D 16x16 f32:     VGPR i -> M = i + 8*(lane>=16), N = lane&15
// B operand stored transposed in LDS (Bt[n][k]) so loads are contiguous.
// ---------------------------------------------------------------------------
static __device__ inline v16bf frag_from_bf16(const bf16_t* base, int stride,
                                              int rowbase, int kbase, int lane) {
  const int r  = rowbase + (lane & 15);
  const int kh = (lane >> 4) << 3;
  const bf16_t* p = base + r * stride + kbase + kh;
  union { uint4 q[2]; v16bf v; } u;
  u.q[0] = *(const uint4*)(p);
  u.q[1] = *(const uint4*)(p + 16);
  return u.v;
}

static __device__ inline v16bf frag_from_f32(const float* base, int stride,
                                             int rowbase, int kbase, int lane,
                                             float scale) {
  const int r  = rowbase + (lane & 15);
  const int kh = (lane >> 4) << 3;
  const float* p = base + r * stride + kbase + kh;
  v16bf o;
#pragma unroll
  for (int i = 0; i < 8; ++i) o[i]     = (bf16_t)(p[i]      * scale);
#pragma unroll
  for (int i = 0; i < 8; ++i) o[i + 8] = (bf16_t)(p[i + 16] * scale);
  return o;
}

static __device__ inline v8f wmma_bf16(v16bf a, v16bf b, v8f c) {
  return __builtin_amdgcn_wmma_f32_16x16x32_bf16(false, a, false, b,
                                                 (short)0, c, false, false);
}

static __device__ inline void acc_store_lds(float* base, int stride, int rowbase,
                                            int colbase, int lane, v8f acc, float mul) {
  const int col = colbase + (lane & 15);
  const int r0  = rowbase + ((lane >> 4) << 3);
#pragma unroll
  for (int i = 0; i < 8; ++i) base[(r0 + i) * stride + col] = acc[i] * mul;
}

// C[M][N] (f32 LDS) = outMul * (diag(rowScale) * A[M][K]) @ Bt[N][K]^T
static __device__ inline void block_mm(float* C, int sC, const float* A, int sA,
                                       const float* Bt, int sB, int M, int N, int K,
                                       const float* rowScale, float outMul,
                                       int wid, int lane) {
  const int ntN = N >> 4;
  const int nt  = (M >> 4) * ntN;
  for (int st = wid; st < nt; st += 8) {
    const int mi = (st / ntN) << 4;
    const int ni = (st % ntN) << 4;
    const float sc = rowScale ? rowScale[mi + (lane & 15)] : 1.0f;
    v8f acc = {0.f, 0.f, 0.f, 0.f, 0.f, 0.f, 0.f, 0.f};
    for (int kk = 0; kk < K; kk += 32) {
      v16bf a = frag_from_f32(A, sA, mi, kk, lane, sc);
      v16bf b = frag_from_f32(Bt, sB, ni, kk, lane, 1.0f);
      acc = wmma_bf16(a, b, acc);
    }
    acc_store_lds(C, sC, mi, ni, lane, acc, outMul);
  }
}

// ---------------------------------------------------------------------------
// Generic tiled WMMA GEMM: C[M,N] = A[M,K] @ B[K,N], fp32 in/out, bf16 compute.
// N, K are template constants -> immediate-offset stores, strength-reduced
// addressing. Block 256 thr (8 waves). Tile 128x128x32; wave covers 64x32.
// A tile staged f32 in LDS via CDNA5 async-to-LDS; B staged bf16 transposed.
// ---------------------------------------------------------------------------
template <int N, int K>
__global__ __launch_bounds__(256) void wmma_gemm_kernel(
    const float* __restrict__ A, const float* __restrict__ Bm,
    float* __restrict__ C) {
  __shared__ float  As[128 * 32];  // f32 staging (async target)
  __shared__ bf16_t Bs[128 * 32];  // transposed: Bs[n][k]
  const int tid = threadIdx.x, lane = tid & 31, wid = tid >> 5;
  const int tm = blockIdx.y * 128, tn = blockIdx.x * 128;
  const int wm = (wid >> 2) * 64, wn = (wid & 3) * 32;

  v8f acc[4][2];
  const v8f vzero = {0.f, 0.f, 0.f, 0.f, 0.f, 0.f, 0.f, 0.f};
#pragma unroll
  for (int m = 0; m < 4; ++m)
#pragma unroll
    for (int n = 0; n < 2; ++n) acc[m][n] = vzero;

  for (int k0 = 0; k0 < K; k0 += 32) {
#pragma unroll
    for (int i = 0; i < 4; ++i) {  // A tile 128x32: async global->LDS (b128)
      const int idx = tid + i * 256;
      const int row = idx >> 3, kk = (idx & 7) << 2;
      async_b128(&As[row * 32 + kk], A + (size_t)(tm + row) * K + k0 + kk);
    }
#pragma unroll
    for (int i = 0; i < 4; ++i) {  // B tile 32x128 -> bf16 transposed store
      const int idx = tid + i * 256;
      const int kk = idx >> 5, col = (idx & 31) << 2;
      float4 f = *(const float4*)(Bm + (size_t)(k0 + kk) * N + tn + col);
      Bs[(col + 0) * 32 + kk] = (bf16_t)f.x;
      Bs[(col + 1) * 32 + kk] = (bf16_t)f.y;
      Bs[(col + 2) * 32 + kk] = (bf16_t)f.z;
      Bs[(col + 3) * 32 + kk] = (bf16_t)f.w;
    }
    if (k0 + 32 < K) {  // prefetch next K-tile (global_prefetch_b8)
      __builtin_prefetch(A + (size_t)(tm + (tid >> 1)) * K + k0 + 32, 0, 1);
      __builtin_prefetch(Bm + (size_t)(k0 + 32 + (tid >> 4)) * N + tn, 0, 1);
    }
    async_wait();
    __syncthreads();

    v16bf af[4], bfg[2];
#pragma unroll
    for (int m = 0; m < 4; ++m)
      af[m] = frag_from_f32(As, 32, wm + m * 16, 0, lane, 1.0f);
#pragma unroll
    for (int n = 0; n < 2; ++n)
      bfg[n] = frag_from_bf16(Bs, 32, wn + n * 16, 0, lane);
#pragma unroll
    for (int m = 0; m < 4; ++m)
#pragma unroll
      for (int n = 0; n < 2; ++n) acc[m][n] = wmma_bf16(af[m], bfg[n], acc[m][n]);
    __syncthreads();
  }

#pragma unroll
  for (int m = 0; m < 4; ++m)
#pragma unroll
    for (int n = 0; n < 2; ++n) {
      const int col = tn + wn + n * 16 + (lane & 15);
      const int r0  = tm + wm + m * 16 + ((lane >> 4) << 3);
      float* cp = C + (size_t)r0 * N + col;
#pragma unroll
      for (int i = 0; i < 8; ++i) cp[(size_t)i * N] = acc[m][n][i];
    }
}

// ---------------------------------------------------------------------------
// Causal depthwise conv1d (K=4, left pad 3) + SiLU, vectorized by 4 channels.
// ---------------------------------------------------------------------------
static __device__ inline float silu_f(float x) { return x / (1.f + __expf(-x)); }

__global__ __launch_bounds__(256) void conv_silu_kernel(
    const float* __restrict__ in, const float* __restrict__ cw,
    float* __restrict__ out) {
  const long long idx4 = (long long)blockIdx.x * 256 + threadIdx.x;
  const long long total4 = (long long)DN_B * DN_S * DN_CONV / 4;
  if (idx4 >= total4) return;
  const int c4 = (int)(idx4 % (DN_CONV / 4)) * 4;
  const long long bs = idx4 / (DN_CONV / 4);
  const int s = (int)(bs % DN_S);
  const long long b = bs / DN_S;

  float w[4][4];
#pragma unroll
  for (int l = 0; l < 4; ++l)
#pragma unroll
    for (int j = 0; j < 4; ++j) w[l][j] = cw[(c4 + l) * 4 + j];

  float a0 = 0.f, a1 = 0.f, a2 = 0.f, a3 = 0.f;
#pragma unroll
  for (int j = 0; j < 4; ++j) {
    const int sj = s + j - 3;
    if (sj < 0) continue;
    const float4 x = *(const float4*)(in + ((b * DN_S + sj) * DN_CONV) + c4);
    a0 += w[0][j] * x.x; a1 += w[1][j] * x.y;
    a2 += w[2][j] * x.z; a3 += w[3][j] * x.w;
  }
  float4 o;
  o.x = silu_f(a0); o.y = silu_f(a1); o.z = silu_f(a2); o.w = silu_f(a3);
  *(float4*)(out + (bs * DN_CONV) + c4) = o;
}

// ---------------------------------------------------------------------------
// beta = sigmoid(hs@W_b); g = -exp(A_log)*softplus(hs@W_a + dt_bias)
// One block per (b,s) row; 8 K-segments x 32 heads.
// ---------------------------------------------------------------------------
__global__ __launch_bounds__(256) void betag_kernel(
    const float* __restrict__ hs, const float* __restrict__ Wb,
    const float* __restrict__ Wa, const float* __restrict__ dtb,
    const float* __restrict__ Alog, float* __restrict__ beta_o,
    float* __restrict__ g_o) {
  __shared__ float row[DN_HID];
  __shared__ float redb[256], reda[256];
  const int bs = blockIdx.x;
  const float* hp = hs + (size_t)bs * DN_HID;
  for (int i = threadIdx.x; i < DN_HID / 4; i += 256)
    *(float4*)&row[i * 4] = *(const float4*)&hp[i * 4];
  __syncthreads();
  const int h = threadIdx.x & 31, seg = threadIdx.x >> 5;
  float pb = 0.f, pa = 0.f;
  for (int k = seg * 256; k < seg * 256 + 256; ++k) {
    const float x = row[k];
    pb += x * Wb[k * DN_HV + h];
    pa += x * Wa[k * DN_HV + h];
  }
  redb[threadIdx.x] = pb; reda[threadIdx.x] = pa;
  __syncthreads();
  if (threadIdx.x < 32) {
    float db = 0.f, da = 0.f;
    for (int s2 = 0; s2 < 8; ++s2) { db += redb[s2 * 32 + h]; da += reda[s2 * 32 + h]; }
    beta_o[(size_t)bs * DN_HV + h] = 1.f / (1.f + __expf(-db));
    const float x = da + dtb[h];
    const float sp = (x > 20.f) ? x : log1pf(__expf(x));
    g_o[(size_t)bs * DN_HV + h] = -__expf(Alog[h]) * sp;
  }
}

// ---------------------------------------------------------------------------
// Chunked gated delta-rule core. One block per (b,h); 8 waves; dynamic LDS.
// State (128x128 f32) lives in WMMA accumulator registers; bf16 shadow in LDS.
// ---------------------------------------------------------------------------
#define LDS_STATE_BF 0
#define LDS_QF       32768
#define LDS_KF       65536
#define LDS_UF       98304
#define LDS_WF       131072
#define LDS_KDT      163840
#define LDS_KBGT     196608
#define LDS_VT       229376
#define LDS_AF       262144
#define LDS_TF       278528
#define LDS_DEC      294912
#define LDS_SMALL    311296
#define DELTA_LDS_BYTES 312576

__global__ __launch_bounds__(256) void delta_core_kernel(
    const float* __restrict__ mixed_conv,  // [B,S,CONV]
    const float* __restrict__ beta_g,      // [B,S,HV]
    const float* __restrict__ g_g,         // [B,S,HV]
    float* __restrict__ core) {            // [B,S,HV,DV]
  extern __shared__ char smem[];
  bf16_t* stateT_bf = (bf16_t*)(smem + LDS_STATE_BF);  // [DV=128][DK=128] bf16
  float* q_f   = (float*)(smem + LDS_QF);    // [64][128]
  float* k_f   = (float*)(smem + LDS_KF);    // [64][128]
  float* u_f   = (float*)(smem + LDS_UF);    // [64][128]
  float* w_f   = (float*)(smem + LDS_WF);    // [64][128]  (holds -w)
  float* kdT_f = (float*)(smem + LDS_KDT);   // [128][64]  k^T * exp(g_last-g)
  float* kbgT_f= (float*)(smem + LDS_KBGT);  // [128][64]  (k*beta*exp(g))^T
  float* vT_f  = (float*)(smem + LDS_VT);    // [128][64]  v_beta^T, then v_new^T
  float* A_f   = (float*)(smem + LDS_AF);    // [64][64]   A, then attn
  float* T_f   = (float*)(smem + LDS_TF);    // [64][64]
  float* dec_f = (float*)(smem + LDS_DEC);   // [64][64]
  float* gcv   = (float*)(smem + LDS_SMALL); // 64
  float* betav = gcv + 64;
  float* rq    = betav + 64;
  float* rk    = rq + 64;
  float* graw  = rk + 64;

  const int tid = threadIdx.x, lane = tid & 31, wid = tid >> 5;
  const int bh = blockIdx.x;
  const int b  = bh >> 5;   // / HV
  const int h  = bh & 31;
  const int hk = h >> 1;    // HV/HK == 2

  v8f stateAcc[8];          // wave owns stateT rows [wid*16, wid*16+16)
  const v8f vzero = {0.f, 0.f, 0.f, 0.f, 0.f, 0.f, 0.f, 0.f};
#pragma unroll
  for (int t = 0; t < 8; ++t) stateAcc[t] = vzero;
  for (int i = tid; i < 128 * 128; i += 256) stateT_bf[i] = (bf16_t)0.0f;
  __syncthreads();

  for (int ch = 0; ch < DN_NC; ++ch) {
    const int c0 = ch * DN_CHUNK;

    // ---- gates: load g,beta; cumsum g -> gc ----
    if (tid < 64) {
      const size_t off = ((size_t)b * DN_S + (c0 + tid)) * DN_HV + h;
      graw[tid]  = g_g[off];
      betav[tid] = beta_g[off];
    }
    __syncthreads();
    if (tid == 0) {
      float a = 0.f;
      for (int i = 0; i < 64; ++i) { a += graw[i]; gcv[i] = a; }
    }
    __syncthreads();
    const float g_last = gcv[63];

    // ---- load q,k raw (async -> LDS); v -> v_beta^T ----
    for (int idx = tid; idx < 64 * 32; idx += 256) {
      const int row = idx >> 5;
      const int c4 = (idx & 31) << 2;
      const size_t ro = ((size_t)b * DN_S + (c0 + row)) * DN_CONV;
      async_b128(&q_f[row * 128 + c4], mixed_conv + ro + hk * DN_DK + c4);
      async_b128(&k_f[row * 128 + c4], mixed_conv + ro + DN_KEYDIM + hk * DN_DK + c4);
      const float4 vv = *(const float4*)(mixed_conv + ro + 2 * DN_KEYDIM + h * DN_DV + c4);
      const float bt = betav[row];
      vT_f[(c4 + 0) * 64 + row] = vv.x * bt;
      vT_f[(c4 + 1) * 64 + row] = vv.y * bt;
      vT_f[(c4 + 2) * 64 + row] = vv.z * bt;
      vT_f[(c4 + 3) * 64 + row] = vv.w * bt;
    }
    async_wait();
    __syncthreads();

    // ---- l2-norm factors ----
    if (tid < 64) {
      float s = 0.f;
      for (int d = 0; d < 128; ++d) { const float x = q_f[tid * 128 + d]; s += x * x; }
      rq[tid] = rsqrtf(s + 1e-6f);
    } else if (tid < 128) {
      const int r = tid - 64;
      float s = 0.f;
      for (int d = 0; d < 128; ++d) { const float x = k_f[r * 128 + d]; s += x * x; }
      rk[r] = rsqrtf(s + 1e-6f);
    }
    __syncthreads();

    // ---- scale q,k; build kdT, kbgT; decay ----
    for (int idx = tid; idx < 64 * 128; idx += 256) {
      const int row = idx >> 7, d = idx & 127;
      const float qn = q_f[idx] * rq[row] * 0.08838834764831845f;  // DK^-0.5
      q_f[idx] = qn;
      const float kn = k_f[idx] * rk[row];
      k_f[idx] = kn;
      kdT_f[d * 64 + row]  = kn * __expf(g_last - gcv[row]);
      kbgT_f[d * 64 + row] = kn * betav[row] * __expf(gcv[row]);
    }
    for (int idx = tid; idx < 64 * 64; idx += 256) {
      const int i = idx >> 6, j = idx & 63;
      dec_f[idx] = (j <= i) ? __expf(gcv[i] - gcv[j]) : 0.f;
    }
    __syncthreads();

    // ---- A = -(k_beta @ k^T) * decay, strictly lower ----
    block_mm(A_f, 64, k_f, 128, k_f, 128, 64, 64, 128, betav, 1.f, wid, lane);
    __syncthreads();
    for (int idx = tid; idx < 4096; idx += 256) {
      const int i = idx >> 6, j = idx & 63;
      A_f[idx] = (j < i) ? (-A_f[idx] * dec_f[idx]) : 0.f;
    }
    __syncthreads();

    // ---- T = inv(I - A) by forward substitution (unit lower) ----
    if (tid < 64) T_f[tid] = (tid == 0) ? 1.f : 0.f;   // row 0
    __syncthreads();
    for (int i = 1; i < 64; ++i) {
      if (tid < 64) {
        float s = (tid == i) ? 1.f : 0.f;
        for (int j = 0; j < i; ++j) s += A_f[i * 64 + j] * T_f[j * 64 + tid];
        T_f[i * 64 + tid] = s;
      }
      __syncthreads();
    }

    // ---- u = T @ v_beta ; w_neg = -(T @ (k_beta*exp(gc))) ----
    block_mm(u_f, 128, T_f, 64, vT_f, 64, 64, 128, 64, nullptr, 1.f, wid, lane);
    block_mm(w_f, 128, T_f, 64, kbgT_f, 64, 64, 128, 64, nullptr, -1.f, wid, lane);
    __syncthreads();

    // ---- attn = (q @ k^T) * decay ----
    block_mm(A_f, 64, q_f, 128, k_f, 128, 64, 64, 128, nullptr, 1.f, wid, lane);
    __syncthreads();
    for (int idx = tid; idx < 4096; idx += 256) A_f[idx] *= dec_f[idx];
    __syncthreads();

    // ---- v_new = u + w_neg @ S ; stored transposed into vT_f ----
    for (int st = wid; st < 32; st += 8) {
      const int mi = (st >> 3) << 4;
      const int ni = (st & 7) << 4;
      const int col = ni + (lane & 15);
      const int r0  = mi + ((lane >> 4) << 3);
      v8f acc;
#pragma unroll
      for (int i = 0; i < 8; ++i) acc[i] = u_f[(r0 + i) * 128 + col];
      for (int kk = 0; kk < 128; kk += 32) {
        v16bf a  = frag_from_f32(w_f, 128, mi, kk, lane, 1.f);
        v16bf bb = frag_from_bf16(stateT_bf, 128, ni, kk, lane);
        acc = wmma_bf16(a, bb, acc);
      }
#pragma unroll
      for (int i = 0; i < 8; ++i) vT_f[col * 64 + (r0 + i)] = acc[i];
    }
    __syncthreads();

    // ---- out = (q*exp(gc)) @ S + attn @ v_new -> global core ----
    for (int st = wid; st < 32; st += 8) {
      const int mi = (st >> 3) << 4;
      const int ni = (st & 7) << 4;
      v8f acc = vzero;
      const float qs = __expf(gcv[mi + (lane & 15)]);
      for (int kk = 0; kk < 128; kk += 32) {
        v16bf a  = frag_from_f32(q_f, 128, mi, kk, lane, qs);
        v16bf bb = frag_from_bf16(stateT_bf, 128, ni, kk, lane);
        acc = wmma_bf16(a, bb, acc);
      }
      for (int kk = 0; kk < 64; kk += 32) {
        v16bf a  = frag_from_f32(A_f, 64, mi, kk, lane, 1.f);
        v16bf bb = frag_from_f32(vT_f, 64, ni, kk, lane, 1.f);
        acc = wmma_bf16(a, bb, acc);
      }
      const int col = ni + (lane & 15);
      const int r0  = mi + ((lane >> 4) << 3);
      float* cp = core + (((size_t)b * DN_S + c0 + r0) * DN_HV + h) * DN_DV + col;
#pragma unroll
      for (int i = 0; i < 8; ++i) cp[(size_t)i * DN_HV * DN_DV] = acc[i];
    }
    __syncthreads();

    // ---- S^T = S^T * exp(g_last) + v_new^T @ kd ----
    const float el = __expf(g_last);
#pragma unroll
    for (int t = 0; t < 8; ++t) {
      const int mi = wid << 4;   // dv subtile (wave-owned)
      const int ni = t << 4;     // dk subtile
      v8f acc = stateAcc[t] * el;
      for (int kk = 0; kk < 64; kk += 32) {
        v16bf a  = frag_from_f32(vT_f, 64, mi, kk, lane, 1.f);
        v16bf bb = frag_from_f32(kdT_f, 64, ni, kk, lane, 1.f);
        acc = wmma_bf16(a, bb, acc);
      }
      stateAcc[t] = acc;
      const int col = ni + (lane & 15);
      const int r0  = mi + ((lane >> 4) << 3);
#pragma unroll
      for (int i = 0; i < 8; ++i)
        stateT_bf[(r0 + i) * 128 + col] = (bf16_t)acc[i];
    }
    __syncthreads();
  }
}

// ---------------------------------------------------------------------------
// Gated RMSNorm over DV + SiLU(z) gate. One wave per (b,s,h).
// ---------------------------------------------------------------------------
__global__ __launch_bounds__(256) void rmsnorm_gate_kernel(
    const float* __restrict__ core, const float* __restrict__ z,
    const float* __restrict__ nw, float* __restrict__ gated) {
  const int lane = threadIdx.x & 31, wid = threadIdx.x >> 5;
  const long long gidx = (long long)blockIdx.x * 8 + wid;
  if (gidx >= (long long)DN_B * DN_S * DN_HV) return;
  const float* cp = core + gidx * DN_DV;
  float vals[4];
  float ss = 0.f;
#pragma unroll
  for (int i = 0; i < 4; ++i) {
    vals[i] = cp[lane + 32 * i];
    ss += vals[i] * vals[i];
  }
#pragma unroll
  for (int off = 16; off > 0; off >>= 1) ss += __shfl_xor(ss, off, 32);
  const float r = rsqrtf(ss * (1.0f / DN_DV) + 1e-6f);
  const int h = (int)(gidx % DN_HV);
  const long long bs = gidx / DN_HV;
  const float* zp = z + bs * DN_VALDIM + (long long)h * DN_DV;
  float* gp = gated + bs * DN_VALDIM + (long long)h * DN_DV;
#pragma unroll
  for (int i = 0; i < 4; ++i) {
    const int dv = lane + 32 * i;
    const float zv = zp[dv];
    gp[dv] = vals[i] * r * nw[dv] * (zv / (1.f + __expf(-zv)));
  }
}

// ---------------------------------------------------------------------------
// kernel_launch
// ---------------------------------------------------------------------------
extern "C" void kernel_launch(void* const* d_in, const int* in_sizes, int n_in,
                              void* d_out, int out_size, void* d_ws, size_t ws_size,
                              hipStream_t stream) {
  (void)in_sizes; (void)n_in; (void)out_size; (void)ws_size;
  const float* hs    = (const float*)d_in[0];
  const float* Wqkv  = (const float*)d_in[1];
  const float* convw = (const float*)d_in[2];
  const float* Wz    = (const float*)d_in[3];
  const float* Wb    = (const float*)d_in[4];
  const float* Wa    = (const float*)d_in[5];
  const float* dtb   = (const float*)d_in[6];
  const float* Alog  = (const float*)d_in[7];
  const float* nw    = (const float*)d_in[8];
  const float* Wout  = (const float*)d_in[9];
  float* out = (float*)d_out;
  float* ws  = (float*)d_ws;

  float* mixed_raw  = ws;                                   // [B,S,CONV]
  float* mixed_conv = mixed_raw  + (size_t)67108864;        // [B,S,CONV]
  float* z_buf      = mixed_conv + (size_t)67108864;        // [B,S,VAL]
  float* beta_buf   = z_buf      + (size_t)33554432;        // [B,S,HV]
  float* g_buf      = beta_buf   + (size_t)262144;          // [B,S,HV]
  float* core_buf   = g_buf      + (size_t)262144;          // [B,S,HV,DV]
  float* gated      = mixed_raw;                            // reuse (conv consumed)

  const int M = DN_B * DN_S;  // 8192

  // 1) mixed_raw = hs @ W_qkv   [8192 x 2048] @ [2048 x 8192]
  wmma_gemm_kernel<DN_CONV, DN_HID>
      <<<dim3(DN_CONV / 128, M / 128), 256, 0, stream>>>(hs, Wqkv, mixed_raw);
  // 1b) z = hs @ W_z            [8192 x 2048] @ [2048 x 4096]
  wmma_gemm_kernel<DN_VALDIM, DN_HID>
      <<<dim3(DN_VALDIM / 128, M / 128), 256, 0, stream>>>(hs, Wz, z_buf);
  // 2) causal depthwise conv + SiLU
  conv_silu_kernel<<<(DN_B * DN_S * DN_CONV / 4) / 256, 256, 0, stream>>>(
      mixed_raw, convw, mixed_conv);
  // 3) beta / g gates
  betag_kernel<<<M, 256, 0, stream>>>(hs, Wb, Wa, dtb, Alog, beta_buf, g_buf);
  // 4) chunked gated delta rule (WMMA core)
  delta_core_kernel<<<DN_B * DN_HV, 256, DELTA_LDS_BYTES, stream>>>(
      mixed_conv, beta_buf, g_buf, core_buf);
  // 5) gated RMSNorm + SiLU(z)
  rmsnorm_gate_kernel<<<(DN_B * DN_S * DN_HV) / 8, 256, 0, stream>>>(
      core_buf, z_buf, nw, gated);
  // 6) out = gated @ W_out      [8192 x 4096] @ [4096 x 2048]
  wmma_gemm_kernel<DN_HID, DN_VALDIM>
      <<<dim3(DN_HID / 128, M / 128), 256, 0, stream>>>(gated, Wout, out);
}
